// QueryInteractionModuleLite_81183471829038
// MI455X (gfx1250) — compile-verified, matrix operand order checked by reading, not run on hardware
//
#include <hip/hip_runtime.h>
#include <hip/hip_bf16.h>
#include <math.h>

// ---------------------------------------------------------------------------
// Problem constants (from reference)
// ---------------------------------------------------------------------------
#define HDIM   256
#define IHDIM  1024
#define NHEADS 8
#define HEADD  32
#define BATCH  16
#define SEQ    1024
#define MROWS  (BATCH * SEQ)        // 16384 tokens
#define QSCALE 0.17677669529663687f // 1/sqrt(32)

typedef __attribute__((ext_vector_type(16))) _Float16 v16h;
typedef __attribute__((ext_vector_type(8)))  _Float16 v8h;
typedef __attribute__((ext_vector_type(8)))  float    v8f;
typedef unsigned int u32x4 __attribute__((ext_vector_type(4)));
typedef int          i32x4 __attribute__((ext_vector_type(4)));
typedef int          i32x8 __attribute__((ext_vector_type(8)));

union U16 { v16h v; v8h h[2]; };

// ---------------------------------------------------------------------------
// Tensor Data Mover: 2-D tile (tile_dw dwords wide x tile_rows) from global
// (row stride stride_dw dwords) into LDS at byte offset lds_off.
// D# packing per CDNA5 ISA (data_size=2 -> 4B).
// ---------------------------------------------------------------------------
#if __has_builtin(__builtin_amdgcn_tensor_load_to_lds)
#define HAVE_TDM 1
__device__ __forceinline__ void tdm_load_2d(unsigned lds_off, const void* gptr,
                                            unsigned tile_dw, unsigned tile_rows,
                                            unsigned stride_dw) {
  unsigned long long ga = (unsigned long long)(uintptr_t)gptr;
  u32x4 g0;
  g0.x = 1u;                                   // count=1, user mode
  g0.y = lds_off;                              // lds_addr (bytes)
  g0.z = (unsigned)ga;                         // global_addr[31:0]
  g0.w = (unsigned)(ga >> 32) | (2u << 30);    // global_addr[56:32] | type=2
  i32x8 g1;
  g1[0] = (int)(2u << 16);                     // wg_mask=0, data_size=2 (4B)
  g1[1] = (int)((tile_dw & 0xFFFFu) << 16);    // tensor_dim0[15:0] (== tile_dw)
  g1[2] = (int)((tile_dw >> 16) | ((tile_rows & 0xFFFFu) << 16)); // d0 hi | d1 lo
  g1[3] = (int)((tile_rows >> 16) | ((tile_dw & 0xFFFFu) << 16)); // d1 hi | tile_dim0
  g1[4] = (int)(tile_rows & 0xFFFFu);          // tile_dim1 | tile_dim2=0
  g1[5] = (int)stride_dw;                      // tensor_dim0_stride[31:0]
  g1[6] = 0;                                   // stride hi | dim1_stride lo
  g1[7] = 0;
  i32x4 z4 = {0, 0, 0, 0};
#if defined(__clang_major__) && __clang_major__ >= 23
  i32x8 z8 = {0, 0, 0, 0, 0, 0, 0, 0};
  __builtin_amdgcn_tensor_load_to_lds(g0, g1, z4, z4, z8, 0);
#else
  __builtin_amdgcn_tensor_load_to_lds(g0, g1, z4, z4, 0);
#endif
}
#endif

// ---------------------------------------------------------------------------
// Weight conversion with transpose: s is R x C (f32, row-major), d is C x R f16
// so GEMM B-fragments load contiguously along K.
// ---------------------------------------------------------------------------
__global__ void cvt_t_kernel(const float* __restrict__ s, _Float16* __restrict__ d,
                             int R, int C) {
  int i = blockIdx.x * blockDim.x + threadIdx.x;
  if (i < R * C) {
    int r = i / C, c = i % C;
    d[(size_t)c * R + r] = (_Float16)s[i];
  }
}

// qk_in = qp + oe (f16), oe_h = oe (f16)
__global__ void prep_kernel(const float* __restrict__ qt, const float* __restrict__ oe,
                            _Float16* __restrict__ qk_h, _Float16* __restrict__ oe_h) {
  int i = blockIdx.x * blockDim.x + threadIdx.x;   // over MROWS*HDIM
  int row = i >> 8, c = i & 255;
  float o = oe[i];
  qk_h[i] = (_Float16)(qt[(size_t)row * 512 + c] + o);
  oe_h[i] = (_Float16)o;
}

// ---------------------------------------------------------------------------
// Generic f16 WMMA GEMM:  out = act((A @ W + bias) * alpha + resid)
// A:  M x K f16 row-major.   Wt: N x K f16 (pre-transposed weight).
// Block tile 64x128 (8 waves, 16x64 strip each, 4 wmma accumulators).
// Double-buffered LDS tiles staged by the Tensor Data Mover: wave 0 issues
// the NEXT tile's two TDM ops, then s_wait_tensorcnt 2 (in-order TENSORcnt
// => current tile landed, next still in flight) so DMA overlaps WMMA.
// vtrans stores out as [b][h][d][seq] for attention V.
// ---------------------------------------------------------------------------
__global__ __launch_bounds__(256) void gemm_f16_kernel(
    const _Float16* __restrict__ A, const _Float16* __restrict__ Wt,
    const float* __restrict__ bias,
    const float* __restrict__ resid, int rstride,
    _Float16* __restrict__ outH, float* __restrict__ outF,
    int M, int K, int N, float alpha, int relu, int vtrans)
{
  __shared__ _Float16 sA[2][64][32];    // double-buffered A tile
  __shared__ _Float16 sB[2][128][32];   // double-buffered Wt tile
  const int tid  = threadIdx.x;
  const int wave = tid >> 5, lane = tid & 31;
  const int half = lane >> 4, l16 = lane & 15;
  const int ntn  = N >> 7;
  const int m0   = (blockIdx.x / ntn) * 64;
  const int n0   = (blockIdx.x % ntn) * 128;
  const int wm   = wave & 3, wn = wave >> 2;

  v8f acc[4] = {};

#if defined(HAVE_TDM)
  if (wave == 0) {  // prologue: stage tile 0 into buffer 0
    tdm_load_2d((unsigned)(uintptr_t)&sA[0][0][0],
                A + (size_t)m0 * K, 16u, 64u, (unsigned)(K >> 1));
    tdm_load_2d((unsigned)(uintptr_t)&sB[0][0][0],
                Wt + (size_t)n0 * K, 16u, 128u, (unsigned)(K >> 1));
  }
#endif

  int ib = 0;
  for (int k0 = 0; k0 < K; k0 += 32, ib ^= 1) {
#if defined(HAVE_TDM)
    if (wave == 0) {
      if (k0 + 32 < K) {  // issue next tile, then wait only for current one
        tdm_load_2d((unsigned)(uintptr_t)&sA[ib ^ 1][0][0],
                    A + (size_t)m0 * K + k0 + 32, 16u, 64u, (unsigned)(K >> 1));
        tdm_load_2d((unsigned)(uintptr_t)&sB[ib ^ 1][0][0],
                    Wt + (size_t)n0 * K + k0 + 32, 16u, 128u, (unsigned)(K >> 1));
        __builtin_amdgcn_s_wait_tensorcnt(2);
      } else {
        __builtin_amdgcn_s_wait_tensorcnt(0);
      }
    }
    __syncthreads();
#else
    {
      const int ar = tid >> 2, ac = (tid & 3) << 3;
      *(v8h*)&sA[ib][ar][ac] = *(const v8h*)(A + (size_t)(m0 + ar) * K + k0 + ac);
      const int br = tid >> 1, bc = (tid & 1) << 4;
      const _Float16* src = Wt + (size_t)(n0 + br) * K + k0 + bc;
      *(v8h*)&sB[ib][br][bc]     = *(const v8h*)(src);
      *(v8h*)&sB[ib][br][bc + 8] = *(const v8h*)(src + 8);
    }
    __syncthreads();
#endif

    // A fragment: two contiguous 16B vector loads per lane
    U16 af;
    {
      const _Float16* pr = &sA[ib][wm * 16 + l16][0];
      af.h[0] = *(const v8h*)(pr + 8 * half);
      af.h[1] = *(const v8h*)(pr + 16 + 8 * half);
    }
    #pragma unroll
    for (int t = 0; t < 4; ++t) {
      U16 bf;
      const _Float16* pc = &sB[ib][wn * 64 + t * 16 + l16][0];
      bf.h[0] = *(const v8h*)(pc + 8 * half);
      bf.h[1] = *(const v8h*)(pc + 16 + 8 * half);
      acc[t] = __builtin_amdgcn_wmma_f32_16x16x32_f16(
          false, af.v, false, bf.v, (short)0, acc[t], false, false);
    }
    __syncthreads();  // all waves done with buf[ib] before it is re-staged
  }

  // fused epilogue: bias, alpha, residual, relu, dual-precision store
  #pragma unroll
  for (int t = 0; t < 4; ++t) {
    const int n = n0 + wn * 64 + t * 16 + l16;
    const float bv = bias ? bias[n] : 0.f;
    #pragma unroll
    for (int i = 0; i < 8; ++i) {
      const int m = m0 + wm * 16 + half * 8 + i;
      float vv = (acc[t][i] + bv) * alpha;
      if (resid) vv += resid[(size_t)m * rstride + n];
      if (relu)  vv = vv > 0.f ? vv : 0.f;
      if (outF) outF[(size_t)m * N + n] = vv;
      if (outH) {
        if (vtrans) { // v: [b][h][d][seq] so attention reads contiguously
          const int bb = m >> 10, srow = m & 1023;
          const int hh = n >> 5,  dd   = n & 31;
          outH[(((size_t)(bb * NHEADS + hh)) * HEADD + dd) * SEQ + srow] = (_Float16)vv;
        } else {
          outH[(size_t)m * N + n] = (_Float16)vv;
        }
      }
    }
  }
}

// ---------------------------------------------------------------------------
// Register-resident flash attention, one wave per 16-query tile.
// Scores computed transposed (A = K-tile, B = Q^T): the score C-layout maps
// in-lane onto the P A-fragment packing -> no LDS staging at all.
// ---------------------------------------------------------------------------
__global__ __launch_bounds__(256) void attn_kernel(
    const _Float16* __restrict__ q, const _Float16* __restrict__ k,
    const _Float16* __restrict__ vt,   // [b][h][d=32][seq]
    const unsigned char* __restrict__ mask,
    _Float16* __restrict__ ctx)
{
  const int tid  = threadIdx.x;
  const int wave = tid >> 5, lane = tid & 31;
  const int half = lane >> 4, l16 = lane & 15;

  const int idx = blockIdx.x;           // B * NH * (SEQ/128) = 1024 blocks
  const int qb  = idx & 7;
  const int h   = (idx >> 3) & 7;
  const int b   = idx >> 6;
  const int q0  = qb * 128 + wave * 16;

  // Q^T B-fragment: lane col = query (l16); two 16B loads
  U16 bq;
  {
    const _Float16* qp = q + ((size_t)(b * SEQ) + q0 + l16) * HDIM + h * HEADD;
    bq.h[0] = *(const v8h*)(qp + 8 * half);
    bq.h[1] = *(const v8h*)(qp + 16 + 8 * half);
  }

  float mrun = -1e30f, lrun = 0.f;
  v8f o0 = {}, o1 = {};
  const unsigned char* mrow = mask + (size_t)b * SEQ;
  const _Float16* vr0 = vt + ((size_t)(b * NHEADS + h) * HEADD + l16) * SEQ;
  const _Float16* vr1 = vr0 + (size_t)16 * SEQ;

  for (int j0 = 0; j0 < SEQ; j0 += 32) {
    float stv[2][8];
    #pragma unroll
    for (int t = 0; t < 2; ++t) {
      U16 ak;
      const _Float16* kp = k + ((size_t)(b * SEQ) + j0 + t * 16 + l16) * HDIM + h * HEADD;
      ak.h[0] = *(const v8h*)(kp + 8 * half);
      ak.h[1] = *(const v8h*)(kp + 16 + 8 * half);
      v8f st = {};
      st = __builtin_amdgcn_wmma_f32_16x16x32_f16(
          false, ak.v, false, bq.v, (short)0, st, false, false);
      #pragma unroll
      for (int i = 0; i < 8; ++i) {
        const int key = j0 + t * 16 + half * 8 + i;
        stv[t][i] = st[i] + (mrow[key] ? 0.f : -1e9f);
      }
    }

    // online softmax; each lane pair (L, L+16) shares one query column
    float tm = stv[0][0];
    #pragma unroll
    for (int t = 0; t < 2; ++t)
      #pragma unroll
      for (int i = 0; i < 8; ++i) tm = fmaxf(tm, stv[t][i]);
    tm = fmaxf(tm, __shfl_xor(tm, 16));
    const float nm = fmaxf(mrun, tm);
    const float es = __expf(mrun - nm);
    float ts = 0.f;
    float pe[2][8];
    #pragma unroll
    for (int t = 0; t < 2; ++t)
      #pragma unroll
      for (int i = 0; i < 8; ++i) {
        const float p = __expf(stv[t][i] - nm);
        pe[t][i] = p; ts += p;
      }
    ts += __shfl_xor(ts, 16);
    lrun = lrun * es + ts;
    mrun = nm;

    // score C-layout -> P A-fragment: pure in-lane element shuffle
    v16h ap;
    #pragma unroll
    for (int i = 0; i < 8; ++i) {
      ap[i]     = (_Float16)pe[0][i];
      ap[8 + i] = (_Float16)pe[1][i];
    }

    // rescale running O rows (row q needs es of lane q)
    #pragma unroll
    for (int i = 0; i < 8; ++i) {
      const float e = __shfl(es, half * 8 + i);
      o0[i] *= e; o1[i] *= e;
    }

    // O += P @ V, V fragments from transposed layout: two 16B loads each
    {
      U16 bv0, bv1;
      bv0.h[0] = *(const v8h*)(vr0 + j0 + 8 * half);
      bv0.h[1] = *(const v8h*)(vr0 + j0 + 16 + 8 * half);
      bv1.h[0] = *(const v8h*)(vr1 + j0 + 8 * half);
      bv1.h[1] = *(const v8h*)(vr1 + j0 + 16 + 8 * half);
      o0 = __builtin_amdgcn_wmma_f32_16x16x32_f16(false, ap, false, bv0.v, (short)0, o0, false, false);
      o1 = __builtin_amdgcn_wmma_f32_16x16x32_f16(false, ap, false, bv1.v, (short)0, o1, false, false);
    }
  }

  // normalize and store ctx (f16)
  #pragma unroll
  for (int i = 0; i < 8; ++i) {
    const float li  = __shfl(lrun, half * 8 + i);
    const float inv = 1.f / li;
    const int qrow  = q0 + half * 8 + i;
    const size_t base = ((size_t)(b * SEQ) + qrow) * HDIM + h * HEADD;
    ctx[base + l16]      = (_Float16)(o0[i] * inv);
    ctx[base + 16 + l16] = (_Float16)(o1[i] * inv);
  }
}

// ---------------------------------------------------------------------------
// LayerNorm over rows of 256, one wave per row (wave32 shfl reductions)
// ---------------------------------------------------------------------------
__global__ __launch_bounds__(256) void ln_kernel(
    const float* __restrict__ x, const float* __restrict__ g, const float* __restrict__ be,
    _Float16* __restrict__ outH, float* __restrict__ outF)
{
  const int wave = threadIdx.x >> 5, lane = threadIdx.x & 31;
  const int row  = blockIdx.x * 8 + wave;
  const float* xr = x + (size_t)row * HDIM;
  float v[8], s = 0.f, sq = 0.f;
  #pragma unroll
  for (int i = 0; i < 8; ++i) { v[i] = xr[lane * 8 + i]; s += v[i]; sq += v[i] * v[i]; }
  #pragma unroll
  for (int m = 16; m >= 1; m >>= 1) { s += __shfl_xor(s, m); sq += __shfl_xor(sq, m); }
  const float mean = s * (1.f / HDIM);
  const float var  = sq * (1.f / HDIM) - mean * mean;
  const float rs   = rsqrtf(var + 1e-5f);
  #pragma unroll
  for (int i = 0; i < 8; ++i) {
    const int c = lane * 8 + i;
    const float y = (v[i] - mean) * rs * g[c] + be[c];
    if (outH) outH[(size_t)row * HDIM + c] = (_Float16)y;
    if (outF) outF[(size_t)row * HDIM + c] = y;
  }
}

// Final LN + active-mask select straight into d_out (row stride 512)
__global__ __launch_bounds__(256) void ln_mask_kernel(
    const float* __restrict__ x, const float* __restrict__ g, const float* __restrict__ be,
    const unsigned char* __restrict__ mask, const float* __restrict__ qt,
    int coloff, float* __restrict__ out)
{
  const int wave = threadIdx.x >> 5, lane = threadIdx.x & 31;
  const int row  = blockIdx.x * 8 + wave;          // token index
  if (!mask[row]) {
    #pragma unroll
    for (int i = 0; i < 8; ++i) {
      const int c = lane * 8 + i;
      out[(size_t)row * 512 + coloff + c] = qt[(size_t)row * 512 + coloff + c];
    }
    return;
  }
  const float* xr = x + (size_t)row * HDIM;
  float v[8], s = 0.f, sq = 0.f;
  #pragma unroll
  for (int i = 0; i < 8; ++i) { v[i] = xr[lane * 8 + i]; s += v[i]; sq += v[i] * v[i]; }
  #pragma unroll
  for (int m = 16; m >= 1; m >>= 1) { s += __shfl_xor(s, m); sq += __shfl_xor(sq, m); }
  const float mean = s * (1.f / HDIM);
  const float var  = sq * (1.f / HDIM) - mean * mean;
  const float rs   = rsqrtf(var + 1e-5f);
  #pragma unroll
  for (int i = 0; i < 8; ++i) {
    const int c = lane * 8 + i;
    out[(size_t)row * 512 + coloff + c] = (v[i] - mean) * rs * g[c] + be[c];
  }
}

// ---------------------------------------------------------------------------
// Host orchestration
// ---------------------------------------------------------------------------
extern "C" void kernel_launch(void* const* d_in, const int* in_sizes, int n_in,
                              void* d_out, int out_size, void* d_ws, size_t ws_size,
                              hipStream_t stream) {
  (void)in_sizes; (void)n_in; (void)out_size; (void)ws_size;
  const float* qt   = (const float*)d_in[0];
  const float* oe   = (const float*)d_in[1];
  const unsigned char* mask = (const unsigned char*)d_in[3];   // jax bool = 1 byte
  const float* Wq  = (const float*)d_in[4];
  const float* Wk  = (const float*)d_in[5];
  const float* Wv  = (const float*)d_in[6];
  const float* bq  = (const float*)d_in[7];
  const float* bk  = (const float*)d_in[8];
  const float* bv  = (const float*)d_in[9];
  const float* Wo  = (const float*)d_in[10];
  const float* bo  = (const float*)d_in[11];
  const float* W1  = (const float*)d_in[12];
  const float* b1  = (const float*)d_in[13];
  const float* W2  = (const float*)d_in[14];
  const float* b2  = (const float*)d_in[15];
  const float* Wp1 = (const float*)d_in[16];
  const float* bp1 = (const float*)d_in[17];
  const float* Wp2 = (const float*)d_in[18];
  const float* bp2 = (const float*)d_in[19];
  const float* Wf1 = (const float*)d_in[20];
  const float* bf1 = (const float*)d_in[21];
  const float* Wf2 = (const float*)d_in[22];
  const float* bf2 = (const float*)d_in[23];
  const float* g1  = (const float*)d_in[24];
  const float* be1 = (const float*)d_in[25];
  const float* g2  = (const float*)d_in[26];
  const float* be2 = (const float*)d_in[27];
  const float* gp  = (const float*)d_in[28];
  const float* bep = (const float*)d_in[29];
  const float* gf  = (const float*)d_in[30];
  const float* bef = (const float*)d_in[31];
  float* out = (float*)d_out;

  // workspace carve-out
  char* ws = (char*)d_ws;
  size_t off = 0;
  auto alloc = [&](size_t bytes) -> char* {
    char* p = ws + off;
    off += (bytes + 255) & ~(size_t)255;
    return p;
  };
  const size_t SMALLW = 256 * 256, BIGW = 256 * 1024;
  _Float16* Wq16  = (_Float16*)alloc(SMALLW * 2);   // all weights transposed: [N][K]
  _Float16* Wk16  = (_Float16*)alloc(SMALLW * 2);
  _Float16* Wv16  = (_Float16*)alloc(SMALLW * 2);
  _Float16* Wo16  = (_Float16*)alloc(SMALLW * 2);
  _Float16* W1h   = (_Float16*)alloc(BIGW * 2);
  _Float16* W2h   = (_Float16*)alloc(BIGW * 2);
  _Float16* Wp1h  = (_Float16*)alloc(BIGW * 2);
  _Float16* Wp2h  = (_Float16*)alloc(BIGW * 2);
  _Float16* Wf1h  = (_Float16*)alloc(BIGW * 2);
  _Float16* Wf2h  = (_Float16*)alloc(BIGW * 2);
  const size_t TOK = (size_t)MROWS * HDIM;          // 4,194,304
  _Float16* qk_h  = (_Float16*)alloc(TOK * 2);
  _Float16* oe_h  = (_Float16*)alloc(TOK * 2);
  _Float16* q_h   = (_Float16*)alloc(TOK * 2);
  _Float16* k_h   = (_Float16*)alloc(TOK * 2);
  _Float16* v_h   = (_Float16*)alloc(TOK * 2);      // stored as [b][h][d][seq]
  _Float16* ctx_h = (_Float16*)alloc(TOK * 2);
  _Float16* t1_h  = (_Float16*)alloc(TOK * 2);
  _Float16* t2_h  = (_Float16*)alloc(TOK * 2);
  float*    t1_f  = (float*)alloc(TOK * 4);
  float*    sbuf  = (float*)alloc(TOK * 4);
  _Float16* hbuf  = (_Float16*)alloc((size_t)MROWS * IHDIM * 2);

  // 1) convert + transpose weights (Wt[N][K])
  cvt_t_kernel<<<(SMALLW + 255) / 256, 256, 0, stream>>>(Wq, Wq16, HDIM, HDIM);
  cvt_t_kernel<<<(SMALLW + 255) / 256, 256, 0, stream>>>(Wk, Wk16, HDIM, HDIM);
  cvt_t_kernel<<<(SMALLW + 255) / 256, 256, 0, stream>>>(Wv, Wv16, HDIM, HDIM);
  cvt_t_kernel<<<(SMALLW + 255) / 256, 256, 0, stream>>>(Wo, Wo16, HDIM, HDIM);
  cvt_t_kernel<<<(BIGW + 255) / 256, 256, 0, stream>>>(W1,  W1h,  HDIM, IHDIM);
  cvt_t_kernel<<<(BIGW + 255) / 256, 256, 0, stream>>>(W2,  W2h,  IHDIM, HDIM);
  cvt_t_kernel<<<(BIGW + 255) / 256, 256, 0, stream>>>(Wp1, Wp1h, HDIM, IHDIM);
  cvt_t_kernel<<<(BIGW + 255) / 256, 256, 0, stream>>>(Wp2, Wp2h, IHDIM, HDIM);
  cvt_t_kernel<<<(BIGW + 255) / 256, 256, 0, stream>>>(Wf1, Wf1h, HDIM, IHDIM);
  cvt_t_kernel<<<(BIGW + 255) / 256, 256, 0, stream>>>(Wf2, Wf2h, IHDIM, HDIM);

  // 2) qk_in / oe f16
  prep_kernel<<<TOK / 256, 256, 0, stream>>>(qt, oe, qk_h, oe_h);

  const int gQKV = (MROWS / 64) * (HDIM / 128);     // 512 blocks
  const int gUP  = (MROWS / 64) * (IHDIM / 128);    // 2048 blocks

  // 3) q/k/v projections (scale folded into q, v stored transposed)
  gemm_f16_kernel<<<gQKV, 256, 0, stream>>>(qk_h, Wq16, bq, nullptr, 0, q_h, nullptr,
                                            MROWS, HDIM, HDIM, QSCALE, 0, 0);
  gemm_f16_kernel<<<gQKV, 256, 0, stream>>>(qk_h, Wk16, bk, nullptr, 0, k_h, nullptr,
                                            MROWS, HDIM, HDIM, 1.f, 0, 0);
  gemm_f16_kernel<<<gQKV, 256, 0, stream>>>(oe_h, Wv16, bv, nullptr, 0, v_h, nullptr,
                                            MROWS, HDIM, HDIM, 1.f, 0, 1);

  // 4) attention
  attn_kernel<<<BATCH * NHEADS * (SEQ / 128), 256, 0, stream>>>(q_h, k_h, v_h, mask, ctx_h);

  // 5) tgt = LN(oe + ctx @ Wo + bo)
  gemm_f16_kernel<<<gQKV, 256, 0, stream>>>(ctx_h, Wo16, bo, oe, HDIM, nullptr, sbuf,
                                            MROWS, HDIM, HDIM, 1.f, 0, 0);
  ln_kernel<<<MROWS / 8, 256, 0, stream>>>(sbuf, g1, be1, t1_h, t1_f);

  // 6) tgt = LN(tgt + relu(tgt@W1+b1)@W2 + b2)
  gemm_f16_kernel<<<gUP, 256, 0, stream>>>(t1_h, W1h, b1, nullptr, 0, hbuf, nullptr,
                                           MROWS, HDIM, IHDIM, 1.f, 1, 0);
  gemm_f16_kernel<<<gQKV, 256, 0, stream>>>(hbuf, W2h, b2, t1_f, HDIM, nullptr, sbuf,
                                            MROWS, IHDIM, HDIM, 1.f, 0, 0);
  ln_kernel<<<MROWS / 8, 256, 0, stream>>>(sbuf, g2, be2, t2_h, nullptr);

  // 7) qp_new = LN(qp + relu(tgt@Wp1+bp1)@Wp2 + bp2)  -> out[:, :256] with mask
  gemm_f16_kernel<<<gUP, 256, 0, stream>>>(t2_h, Wp1h, bp1, nullptr, 0, hbuf, nullptr,
                                           MROWS, HDIM, IHDIM, 1.f, 1, 0);
  gemm_f16_kernel<<<gQKV, 256, 0, stream>>>(hbuf, Wp2h, bp2, qt, 512, nullptr, sbuf,
                                            MROWS, IHDIM, HDIM, 1.f, 0, 0);
  ln_mask_kernel<<<MROWS / 8, 256, 0, stream>>>(sbuf, gp, bep, mask, qt, 0, out);

  // 8) qf_new = LN(qf + relu(tgt@Wf1+bf1)@Wf2 + bf2)  -> out[:, 256:] with mask
  gemm_f16_kernel<<<gUP, 256, 0, stream>>>(t2_h, Wf1h, bf1, nullptr, 0, hbuf, nullptr,
                                           MROWS, HDIM, IHDIM, 1.f, 1, 0);
  gemm_f16_kernel<<<gQKV, 256, 0, stream>>>(hbuf, Wf2h, bf2, qt + 256, 512, nullptr, sbuf,
                                            MROWS, IHDIM, HDIM, 1.f, 0, 0);
  ln_mask_kernel<<<MROWS / 8, 256, 0, stream>>>(sbuf, gf, bef, mask, qt, 256, out);
}
